// ComplEx_Gate_att_49203145342987
// MI455X (gfx1250) — matrix operand order repaired and checked by reading, not compiled, not run on hardware
//
#include <hip/hip_runtime.h>
#include <math.h>

// ---------------------------------------------------------------------------
// ComplEx + literal-attention + gating, restructured for CDNA5 (gfx1250).
// Key algebra: kv = lit[n,a] * emb_att[a]  =>  attention logits are
// lit[n,a]*s[b,h,a] (bk cancels in softmax), o = C @ VAfull + bv, and the
// Wo projection is folded into the gate matrices (M1 = gAtt@Wo, M2 = g2W@Wo).
// Entity embedding is concatenated into the same GEMM:  [o | ent] @ [M | G]^T.
// Heavy path = bf16 WMMA (v_wmma_f32_16x16x32_bf16), fp32 accumulation.
// Both A (LDS) and B (global) operands are stored pre-swizzled in WMMA
// fragment order so fragment fetches are 2x b128 loads per lane.
// ---------------------------------------------------------------------------

typedef unsigned short u16;
typedef unsigned int   u32;
typedef __attribute__((ext_vector_type(4)))  u32   u32x4;
typedef __attribute__((ext_vector_type(16))) __bf16 v16bf;
typedef __attribute__((ext_vector_type(8)))  float  v8f;

union BF16x16 { v16bf v; u16 u[16]; u32x4 q[2]; };

__device__ __forceinline__ u16 f2bf(float f) {          // round-to-nearest-even
  u32 u = __float_as_uint(f);
  u32 r = u + 0x7FFFu + ((u >> 16) & 1u);
  return (u16)(r >> 16);
}

// Inverse of the A-fragment per-lane K mapping (k = i + 8*(i>=8) + 8*(lane>=16)):
// element (row, k) of a 16xK A operand lives at swizzled slot
//   ((k>>5)*32 + g*16 + row)*16 + i , g=(kk>>3)&1, i=(kk&7)+8*(kk>=16), kk=k&31
// so each lane's 16 halves for one K-tile are contiguous (2 x b128).
__device__ __forceinline__ int swzA(int row, int k) {
  int kt = k >> 5, kk = k & 31;
  int g = (kk >> 3) & 1;
  int i = (kk & 7) + ((kk >= 16) ? 8 : 0);
  return ((kt * 32) + (g * 16 + row)) * 16 + i;
}

#define Dd     200
#define ATTd   200
#define Hh     4
#define HEADD  50
#define Nn     10000
#define Aa     40
#define Bb     16
#define KC     160                 // GEMM1 K  (= H*A, exact multiple of 32)
#define K2     416                 // GEMM2 K  (ATT 200 + D 200, padded to 416)
#define CT     13                  // column tiles (208 padded output columns)
#define KT1    5                   // K tiles GEMM1
#define KT2    13                  // K tiles GEMM2
#define MTSZ   (KT2*CT*32*16)      // swizzled bf16 matrix size (86528 elems)
#define VAFSZ  (KT1*CT*32*16)      // 33280 elems
#define SCALEF 0.14142135623730951f

// ---------------------------------------------------------------------------
// prep_a: per complex component -- KA/VA = emb_att@{Wk,Wv}^T, query chain
// q = tanh(rel@Wlin^T+b), qq = q@Wq^T+bq, s[b,h,a] = SCALE * qq_h . KA_h,
// and VAfull (zero-padded, head-block-diagonal) stored pre-swizzled in WMMA
// B-fragment order.
// ---------------------------------------------------------------------------
__global__ __launch_bounds__(256) void prep_a(
    const int* rel, const float* emb_rel, const float* linW, const float* linB,
    const float* emb_att, const float* Wq, const float* bq,
    const float* Wk, const float* Wv,
    float* KA, float* VA, float* S, u16* VAF)
{
  __shared__ float qlin[Bb][ATTd];
  __shared__ float qq[Bb][ATTd];
  const int tid = threadIdx.x;

  for (int idx = tid; idx < Aa * ATTd; idx += 256) {
    int a = idx / ATTd, j = idx % ATTd;
    const float* av = emb_att + a * ATTd;
    float sk = 0.f, sv = 0.f;
    for (int d = 0; d < ATTd; ++d) { float x = av[d]; sk += x * Wk[j*ATTd + d]; sv += x * Wv[j*ATTd + d]; }
    KA[idx] = sk; VA[idx] = sv;
  }
  for (int idx = tid; idx < Bb * ATTd; idx += 256) {
    int b = idx / ATTd, j = idx % ATTd;
    const float* rv = emb_rel + (size_t)rel[b] * Dd;
    float s = linB[j];
    for (int d = 0; d < Dd; ++d) s += rv[d] * linW[j*Dd + d];
    qlin[b][j] = tanhf(s);
  }
  __syncthreads();
  for (int idx = tid; idx < Bb * ATTd; idx += 256) {
    int b = idx / ATTd, k = idx % ATTd;
    float s = bq[k];
    for (int j = 0; j < ATTd; ++j) s += qlin[b][j] * Wq[k*ATTd + j];
    qq[b][k] = s;
  }
  __syncthreads();
  for (int idx = tid; idx < Bb * Hh * Aa; idx += 256) {
    int b = idx / (Hh * Aa); int h = (idx / Aa) % Hh; int a = idx % Aa;
    float s = 0.f;
    for (int t = 0; t < HEADD; ++t) s += qq[b][h*HEADD + t] * KA[a*ATTd + h*HEADD + t];
    S[idx] = s * SCALEF;
  }
  // VAfull[k=h*40+a][d] = (d/50==h) ? VA[a][d] : 0 ; swizzled B-fragment order
  for (int idx = tid; idx < VAFSZ; idx += 256) {
    int i = idx & 15; int L = (idx >> 4) & 31; int rest = idx >> 9;
    int ct = rest % CT; int kt = rest / CT;
    int k = kt*32 + (i & 7) + ((i >= 8) ? 16 : 0) + ((L >= 16) ? 8 : 0);
    int d = ct*16 + (L & 15);
    float val = 0.f;
    if (d < ATTd) { int h = k / Aa, a = k % Aa; if (d / HEADD == h) val = VA[a*ATTd + d]; }
    VAF[idx] = f2bf(val);
  }
}

// ---------------------------------------------------------------------------
// prep_b: fused gate matrices, stored pre-swizzled bf16.
//   mat = 2*comp + which;  which==0 -> [gAtt@Wo | gW_d]^T (g_emb path)
//                          which==1 -> [g2W @Wo | g1W ]^T (gate path)
// plus constants c1 = gAtt@bo + gb,  c2 = g2W@bo + gate_bias.
// ---------------------------------------------------------------------------
struct BParams {
  const float* gW[2]; const float* g1W[2]; const float* g2W[2];
  const float* gb[2]; const float* gbias[2];
  const float* Wo[2]; const float* bo[2];
  u16* MT; float* C1[2]; float* C2[2];
};

__global__ __launch_bounds__(256) void prep_b(BParams P)
{
  const int total = 4 * MTSZ + 2 * 2 * Dd;
  for (int idx = blockIdx.x * 256 + threadIdx.x; idx < total; idx += gridDim.x * 256) {
    if (idx < 4 * MTSZ) {
      int mat = idx / MTSZ; int w = idx % MTSZ;
      int comp = mat >> 1; int which = mat & 1;
      int i = w & 15; int L = (w >> 4) & 31; int rest = w >> 9;
      int ct = rest % CT; int kt = rest / CT;
      int k = kt*32 + (i & 7) + ((i >= 8) ? 16 : 0) + ((L >= 16) ? 8 : 0);
      int d = ct*16 + (L & 15);
      float val = 0.f;
      if (d < Dd && k < 400) {
        if (k < ATTd) {
          const float* Wo = P.Wo[comp];
          const float* g = (which == 0) ? (P.gW[comp] + (size_t)d*400 + Dd)
                                        : (P.g2W[comp] + (size_t)d*ATTd);
          float s = 0.f;
          for (int j = 0; j < ATTd; ++j) s += g[j] * Wo[j*ATTd + k];
          val = s;
        } else {
          int dd = k - ATTd;
          val = (which == 0) ? P.gW[comp][(size_t)d*400 + dd]
                             : P.g1W[comp][(size_t)d*Dd + dd];
        }
      }
      P.MT[idx] = f2bf(val);
    } else {
      int e = idx - 4 * MTSZ;
      int comp = e / (2 * Dd); int which = (e / Dd) % 2; int d = e % Dd;
      if (which == 0) {
        const float* g = P.gW[comp] + (size_t)d*400 + Dd;
        float s = P.gb[comp][d];
        for (int j = 0; j < ATTd; ++j) s += g[j] * P.bo[comp][j];
        P.C1[comp][d] = s;
      } else {
        const float* g = P.g2W[comp] + (size_t)d*ATTd;
        float s = P.gbias[comp][d];
        for (int j = 0; j < ATTd; ++j) s += g[j] * P.bo[comp][j];
        P.C2[comp][d] = s;
      }
    }
  }
}

// ---------------------------------------------------------------------------
// prep_d: e1 path (B=16 rows, pure VALU). _attn_e1 == _attn_e2 at n=e1[b].
// Produces pr = e1r_g*relr - e1i_g*reli ; pi = e1r_g*reli + e1i_g*relr.
// ---------------------------------------------------------------------------
struct DParams {
  const int* e1; const int* rel; const float* lit;
  const float* emb_e[2]; const float* emb_rel[2];
  const float* S[2]; const float* VA[2];
  const float* Wo[2]; const float* bo[2]; const float* bv[2];
  const float* gW[2]; const float* gb[2]; const float* g1W[2];
  const float* g2W[2]; const float* gbias[2];                 // gate_e1 params
  float* P0; float* P1;
};

__global__ __launch_bounds__(256) void prep_d(DParams P)
{
  __shared__ float cc[Hh*Aa];
  __shared__ float ob[ATTd];
  __shared__ float attb[ATTd];
  __shared__ float e1v[Dd];
  __shared__ float e1g[2][Dd];
  const int tid = threadIdx.x;
  for (int b = 0; b < Bb; ++b) {
    const int n = P.e1[b];
    for (int comp = 0; comp < 2; ++comp) {
      if (tid < Hh) {
        int h = tid;
        const float* sp = P.S[comp] + b*Hh*Aa + h*Aa;
        const float* lr = P.lit + (size_t)n*Aa;
        float m = -1e30f;
        for (int a = 0; a < Aa; ++a) m = fmaxf(m, lr[a]*sp[a]);
        float sum = 0.f;
        for (int a = 0; a < Aa; ++a) sum += __expf(lr[a]*sp[a] - m);
        float inv = 1.f / sum;
        for (int a = 0; a < Aa; ++a) cc[h*Aa + a] = __expf(lr[a]*sp[a] - m) * inv * lr[a];
      }
      if (tid < Dd) e1v[tid] = P.emb_e[comp][(size_t)n*Dd + tid];
      __syncthreads();
      if (tid < ATTd) {
        int h = tid / HEADD;
        float s = P.bv[comp][tid];
        for (int a = 0; a < Aa; ++a) s += cc[h*Aa + a] * P.VA[comp][a*ATTd + tid];
        ob[tid] = s;
      }
      __syncthreads();
      if (tid < ATTd) {
        float s = P.bo[comp][tid];
        for (int j = 0; j < ATTd; ++j) s += ob[j] * P.Wo[comp][tid*ATTd + j];
        attb[tid] = s;
      }
      __syncthreads();
      if (tid < Dd) {
        int d = tid;
        const float* gWr = P.gW[comp] + (size_t)d*400;
        float ga = P.gb[comp][d];
        for (int dd = 0; dd < Dd; ++dd) ga += e1v[dd] * gWr[dd];
        for (int k = 0; k < ATTd; ++k)  ga += attb[k] * gWr[Dd + k];
        float gemb = tanhf(ga);
        const float* g1 = P.g1W[comp] + (size_t)d*Dd;
        const float* g2 = P.g2W[comp] + (size_t)d*ATTd;
        float ta = P.gbias[comp][d];
        for (int dd = 0; dd < Dd; ++dd) ta += e1v[dd] * g1[dd];
        for (int k = 0; k < ATTd; ++k)  ta += attb[k] * g2[k];
        float gt = 1.f / (1.f + __expf(-ta));
        e1g[comp][d] = (1.f - gt) * e1v[d] + gt * gemb;
      }
      __syncthreads();
    }
    if (tid < Dd) {
      int r = P.rel[b];
      float rr = P.emb_rel[0][(size_t)r*Dd + tid];
      float ri = P.emb_rel[1][(size_t)r*Dd + tid];
      P.P0[b*Dd + tid] = e1g[0][tid]*rr - e1g[1][tid]*ri;
      P.P1[b*Dd + tid] = e1g[0][tid]*ri + e1g[1][tid]*rr;
    }
    __syncthreads();
  }
}

// ---------------------------------------------------------------------------
// score_kernel: one wave per (16-entity tile, b).
//   softmax -> C (swizzled LDS) -> WMMA GEMM1 -> O||ent (swizzled LDS)
//   -> dual WMMA GEMM2 -> fused tanh/sigmoid gate + ComplEx dot epilogue
//   -> shfl width-16 reduction -> sigmoid output.
// A fragments: 2x ds_load_b128 ; B fragments: 2x global_load_b128.
// ---------------------------------------------------------------------------
struct EParams {
  const float* lit; const float* emb_e[2]; const float* bv[2];
  const float* S[2]; const u16* VAF[2]; const u16* MT1[2]; const u16* MT2[2];
  const float* C1[2]; const float* C2[2]; const float* P[2];
  const float* bias; float* out;
};

__global__ __launch_bounds__(32) void score_kernel(EParams P)
{
  __shared__ float litS[16][Aa];
  __shared__ float sS[Hh*Aa];
  __shared__ __align__(32) u16 CmS[KT1 * 32 * 16];   // C, A-fragment order
  __shared__ __align__(32) u16 OmS[KT2 * 32 * 16];   // O||ent, A-fragment order
  const int lane = threadIdx.x;
  const int b  = blockIdx.y;
  const int n0 = blockIdx.x * 16;
  const int half = lane >> 4;
  const int col  = lane & 15;

  __builtin_prefetch(P.MT1[0], 0, 1);   // global_prefetch: L2-resident B mats
  __builtin_prefetch(P.MT2[0], 0, 1);

  for (int idx = lane; idx < 16 * Aa; idx += 32)
    litS[idx / Aa][idx % Aa] = P.lit[(size_t)(n0 + idx / Aa) * Aa + idx % Aa];

  float score[8];
#pragma unroll
  for (int r = 0; r < 8; ++r) score[r] = 0.f;

  for (int comp = 0; comp < 2; ++comp) {
    for (int idx = lane; idx < Hh * Aa; idx += 32) sS[idx] = P.S[comp][b*Hh*Aa + idx];
    __syncthreads();

    // softmax coefficients c = softmax_a(lit*s) * lit  (row = col; 2 heads/lane)
    {
      const int row = col;
      for (int hh = 0; hh < 2; ++hh) {
        int h = half*2 + hh;
        const float* sp = &sS[h*Aa];
        float m = -1e30f;
        for (int a = 0; a < Aa; ++a) m = fmaxf(m, litS[row][a]*sp[a]);
        float sum = 0.f;
        for (int a = 0; a < Aa; ++a) sum += __expf(litS[row][a]*sp[a] - m);
        float inv = 1.f / sum;
        for (int a = 0; a < Aa; ++a)
          CmS[swzA(row, h*Aa + a)] = f2bf(__expf(litS[row][a]*sp[a] - m) * inv * litS[row][a]);
      }
    }
    __syncthreads();

    // GEMM1: O = C @ VAfull  (+ bv)
    const u32x4* vaf = (const u32x4*)P.VAF[comp];
    const float* bvp = P.bv[comp];
    for (int ct = 0; ct < CT; ++ct) {
      v8f acc;
#pragma unroll
      for (int r = 0; r < 8; ++r) acc[r] = 0.f;
      for (int kt = 0; kt < KT1; ++kt) {
        BF16x16 Af;
        const u32x4* ap = (const u32x4*)&CmS[(kt*32 + lane) * 16];
        Af.q[0] = ap[0]; Af.q[1] = ap[1];
        BF16x16 Bf;
        const u32x4* bp = vaf + ((size_t)((kt*CT + ct)*32 + lane) * 2);
        Bf.q[0] = bp[0]; Bf.q[1] = bp[1];
        acc = __builtin_amdgcn_wmma_f32_16x16x32_bf16(false, Af.v, false, Bf.v,
                                                      (short)0, acc, false, false);
      }
      int d = ct*16 + col;
      if (d < ATTd) {
#pragma unroll
        for (int r = 0; r < 8; ++r) OmS[swzA(r + half*8, d)] = f2bf(acc[r] + bvp[d]);
      }
    }
    // entity part + zero pad (columns 200..415)
    const float* ee = P.emb_e[comp];
    for (int idx = lane; idx < 16 * 216; idx += 32) {
      int m = idx / 216; int c2 = 200 + idx % 216;
      OmS[swzA(m, c2)] = (c2 < 400) ? f2bf(ee[(size_t)(n0 + m)*Dd + (c2 - 200)]) : (u16)0;
    }
    __syncthreads();

    // GEMM2 (dual) + fused gate/scoring epilogue
    const u32x4* mt1 = (const u32x4*)P.MT1[comp];
    const u32x4* mt2 = (const u32x4*)P.MT2[comp];
    const float* c1p = P.C1[comp];
    const float* c2p = P.C2[comp];
    const float* pv  = P.P[comp] + b * Dd;
    for (int ct = 0; ct < CT; ++ct) {
      v8f a1, a2;
#pragma unroll
      for (int r = 0; r < 8; ++r) { a1[r] = 0.f; a2[r] = 0.f; }
      for (int kt = 0; kt < KT2; ++kt) {
        BF16x16 Af;
        const u32x4* ap = (const u32x4*)&OmS[(kt*32 + lane) * 16];
        Af.q[0] = ap[0]; Af.q[1] = ap[1];
        size_t fo = (size_t)((kt*CT + ct)*32 + lane) * 2;
        BF16x16 B1; B1.q[0] = mt1[fo]; B1.q[1] = mt1[fo + 1];
        BF16x16 B2; B2.q[0] = mt2[fo]; B2.q[1] = mt2[fo + 1];
        a1 = __builtin_amdgcn_wmma_f32_16x16x32_bf16(false, Af.v, false, B1.v,
                                                     (short)0, a1, false, false);
        a2 = __builtin_amdgcn_wmma_f32_16x16x32_bf16(false, Af.v, false, B2.v,
                                                     (short)0, a2, false, false);
      }
      int d = ct*16 + col;
      if (d < Dd) {
        float c1v = c1p[d], c2v = c2p[d], pd = pv[d];
#pragma unroll
        for (int r = 0; r < 8; ++r) {
          int m = r + half*8;
          float entv = ee[(size_t)(n0 + m)*Dd + d];
          float gemb = tanhf(a1[r] + c1v);
          float gt   = 1.f / (1.f + __expf(-(a2[r] + c2v)));
          score[r] += pd * ((1.f - gt)*entv + gt*gemb);
        }
      }
    }
    __syncthreads();
  }

  // width-16 reduction across the N columns of each half, then sigmoid output
#pragma unroll
  for (int r = 0; r < 8; ++r) {
    float v = score[r];
    for (int off = 1; off < 16; off <<= 1) v += __shfl_xor(v, off, 16);
    if (col == 0) {
      int n = n0 + r + half*8;
      P.out[(size_t)b * Nn + n] = 1.f / (1.f + __expf(-(v + P.bias[n])));
    }
  }
}

// ---------------------------------------------------------------------------
extern "C" void kernel_launch(void* const* d_in, const int* in_sizes, int n_in,
                              void* d_out, int out_size, void* d_ws, size_t ws_size,
                              hipStream_t stream)
{
  (void)in_sizes; (void)n_in; (void)out_size; (void)ws_size;
  const int* e1  = (const int*)d_in[0];
  const int* rel = (const int*)d_in[1];
  const float* emb_e[2]   = { (const float*)d_in[2], (const float*)d_in[3] };
  const float* emb_rel[2] = { (const float*)d_in[4], (const float*)d_in[5] };
  const float* emb_att[2] = { (const float*)d_in[6], (const float*)d_in[7] };
  const float* lit = (const float*)d_in[8];
  const float* linW[2] = { (const float*)d_in[9],  (const float*)d_in[11] };
  const float* linB[2] = { (const float*)d_in[10], (const float*)d_in[12] };
  auto mha = [&](int c, int k) { return (const float*)d_in[13 + 8*c + k]; }; // Wq,Wk,Wv,Wo,bq,bk,bv,bo
  auto gte = [&](int base, int c, int k) { return (const float*)d_in[base + 5*c + k]; }; // gW,gb,g1W,g2W,gate_bias
  const float* bias = (const float*)d_in[49];
  float* out = (float*)d_out;

  // workspace carve (256B aligned slabs; ~1 MB total)
  char* w = (char*)d_ws;
  size_t off = 0;
  auto carve = [&](size_t bytes) -> void* {
    void* p = w + off; off += (bytes + 255) & ~(size_t)255; return p;
  };
  float* Sb[2];  float* VAb[2];  u16* VAFb[2];
  for (int c = 0; c < 2; ++c) {
    Sb[c]   = (float*)carve(Bb*Hh*Aa * sizeof(float));
    VAb[c]  = (float*)carve(Aa*ATTd * sizeof(float));
    VAFb[c] = (u16*)  carve((size_t)VAFSZ * sizeof(u16));
  }
  u16* MTb = (u16*)carve((size_t)4 * MTSZ * sizeof(u16));
  float* C1b[2]; float* C2b[2];
  for (int c = 0; c < 2; ++c) {
    C1b[c] = (float*)carve(Dd * sizeof(float));
    C2b[c] = (float*)carve(Dd * sizeof(float));
  }
  float* P0b = (float*)carve(Bb*Dd * sizeof(float));
  float* P1b = (float*)carve(Bb*Dd * sizeof(float));
  float* KAb = (float*)carve(Aa*ATTd * sizeof(float));

  // 1) per-component query/attention precompute
  for (int c = 0; c < 2; ++c)
    prep_a<<<1, 256, 0, stream>>>(rel, emb_rel[c], linW[c], linB[c],
                                  emb_att[c], mha(c,0), mha(c,4), mha(c,1), mha(c,2),
                                  KAb, VAb[c], Sb[c], VAFb[c]);

  // 2) fused gate matrices (gate_e2 params, Wo folded)
  BParams BP;
  for (int c = 0; c < 2; ++c) {
    BP.gW[c] = gte(39,c,0); BP.gb[c] = gte(39,c,1); BP.g1W[c] = gte(39,c,2);
    BP.g2W[c] = gte(39,c,3); BP.gbias[c] = gte(39,c,4);
    BP.Wo[c] = mha(c,3); BP.bo[c] = mha(c,7);
    BP.C1[c] = C1b[c]; BP.C2[c] = C2b[c];
  }
  BP.MT = MTb;
  prep_b<<<1356, 256, 0, stream>>>(BP);

  // 3) e1 path -> pr/pi (gate_e1 params)
  DParams DP;
  DP.e1 = e1; DP.rel = rel; DP.lit = lit;
  for (int c = 0; c < 2; ++c) {
    DP.emb_e[c] = emb_e[c]; DP.emb_rel[c] = emb_rel[c];
    DP.S[c] = Sb[c]; DP.VA[c] = VAb[c];
    DP.Wo[c] = mha(c,3); DP.bo[c] = mha(c,7); DP.bv[c] = mha(c,6);
    DP.gW[c] = gte(29,c,0); DP.gb[c] = gte(29,c,1); DP.g1W[c] = gte(29,c,2);
    DP.g2W[c] = gte(29,c,3); DP.gbias[c] = gte(29,c,4);
  }
  DP.P0 = P0b; DP.P1 = P1b;
  prep_d<<<1, 256, 0, stream>>>(DP);

  // 4) main fused WMMA kernel over all (b, entity-tile) pairs
  EParams EP;
  EP.lit = lit; EP.bias = bias; EP.out = out;
  for (int c = 0; c < 2; ++c) {
    EP.emb_e[c] = emb_e[c]; EP.bv[c] = mha(c,6);
    EP.S[c] = Sb[c]; EP.VAF[c] = VAFb[c];
    EP.MT1[c] = MTb + (size_t)(2*c + 0) * MTSZ;
    EP.MT2[c] = MTb + (size_t)(2*c + 1) * MTSZ;
    EP.C1[c] = C1b[c]; EP.C2[c] = C2b[c];
  }
  EP.P[0] = P0b; EP.P[1] = P1b;
  score_kernel<<<dim3(Nn/16, Bb, 1), dim3(32, 1, 1), 0, stream>>>(EP);
}
